// GPT_44452911514393
// MI455X (gfx1250) — matrix operand order repaired and checked by reading, not compiled, not run on hardware
//
#include <hip/hip_runtime.h>
#include <hip/hip_bf16.h>
#include <math.h>

// ---------------- model constants ----------------
constexpr int Bc  = 2;
constexpr int Tc  = 1024;
constexpr int Ec  = 2048;
constexpr int Hc  = 16;
constexpr int Gc  = 4;
constexpr int HSc = 128;
constexpr int Lc  = 4;
constexpr int Vc  = 32000;
constexpr int FFc = 8192;
constexpr int Mr  = Bc * Tc;                 // 2048 rows
constexpr int QKVO = (Hc + 2 * Gc) * HSc;    // 3072

typedef __bf16 bf16;
typedef __attribute__((ext_vector_type(16))) __bf16 v16bf;
typedef __attribute__((ext_vector_type(8)))  __bf16 v8bf;
typedef __attribute__((ext_vector_type(8)))  float  v8f;

#define WMMA_BF16(a, b, c) \
    __builtin_amdgcn_wmma_f32_16x16x32_bf16(false, (a), false, (b), (short)0, (c), false, false)

// A-matrix 16x32 bf16 fragment: lanes 0-15 hold K 0..15, lanes 16-31 hold
// K 16..31, for row M = lane%16. One contiguous 32B chunk per lane.
__device__ __forceinline__ v16bf load_a_frag(const bf16* row, int lane, int k) {
    return *(const v16bf*)(row + k + ((lane >> 4) & 1) * 16);
}

// B-matrix 32x16 bf16 fragment: lanes 0-15 hold K 0..15, lanes 16-31 hold
// K 16..31, for column N = lane%16. One contiguous 32B chunk per lane.
__device__ __forceinline__ v16bf load_b_frag(const bf16* row, int lane, int k) {
    return *(const v16bf*)(row + k + ((lane >> 4) & 1) * 16);
}

// ---------------- elementwise kernels ----------------
__global__ void f32_to_bf16_kernel(const float* __restrict__ src,
                                   bf16* __restrict__ dst, long long n) {
    long long i = (long long)blockIdx.x * blockDim.x + threadIdx.x;
    long long stride = (long long)gridDim.x * blockDim.x;
    for (; i < n; i += stride) dst[i] = (bf16)src[i];
}

__global__ void embed_kernel(const int* __restrict__ idx,
                             const float* __restrict__ wte,
                             float* __restrict__ x) {
    const int row = blockIdx.x;
    const int tok = idx[row];
    const float* src = wte + (size_t)tok * Ec;
    float* d = x + (size_t)row * Ec;
    for (int e = threadIdx.x; e < Ec; e += blockDim.x) d[e] = src[e];
}

__global__ __launch_bounds__(256)
void rmsnorm_kernel(const float* __restrict__ x, const float* __restrict__ w,
                    bf16* __restrict__ out) {
    __shared__ float red[256];
    const int row = blockIdx.x;
    const float* xr = x + (size_t)row * Ec;
    float s = 0.f;
    for (int e = threadIdx.x; e < Ec; e += 256) { float v = xr[e]; s += v * v; }
    red[threadIdx.x] = s;
    __syncthreads();
    for (int off = 128; off > 0; off >>= 1) {
        if (threadIdx.x < off) red[threadIdx.x] += red[threadIdx.x + off];
        __syncthreads();
    }
    const float rinv = rsqrtf(red[0] / (float)Ec + 1e-5f);
    bf16* o = out + (size_t)row * Ec;
    for (int e = threadIdx.x; e < Ec; e += 256) o[e] = (bf16)(xr[e] * rinv * w[e]);
}

// RoPE on q (H heads) and k (G heads). qkv layout per row: [q 16*128 | k 4*128 | v 4*128]
// Writes q bf16 (B,H,T,HS) and k bf16 (B,G,T,HS).
__global__ void rope_qk_kernel(const float* __restrict__ qkv,
                               bf16* __restrict__ q, bf16* __restrict__ k) {
    int tid = blockIdx.x * blockDim.x + threadIdx.x;
    const int i = tid & 63;           // rotation pair index 0..63
    int rest = tid >> 6;
    const int head = rest % (Hc + Gc);
    rest /= (Hc + Gc);
    const int t = rest % Tc;
    const int b = rest / Tc;
    if (b >= Bc) return;
    const float ang = (float)t * powf(10000.0f, -(float)i / 64.0f);
    const float c = cosf(ang), s = sinf(ang);
    const float* row = qkv + (size_t)(b * Tc + t) * QKVO;
    if (head < Hc) {
        const float x1 = row[head * HSc + i];
        const float x2 = row[head * HSc + i + 64];
        size_t o = (((size_t)b * Hc + head) * Tc + t) * HSc + i;
        q[o]      = (bf16)(x1 * c - x2 * s);
        q[o + 64] = (bf16)(x2 * c + x1 * s);
    } else {
        const int g = head - Hc;
        const float x1 = row[Hc * HSc + g * HSc + i];
        const float x2 = row[Hc * HSc + g * HSc + i + 64];
        size_t o = (((size_t)b * Gc + g) * Tc + t) * HSc + i;
        k[o]      = (bf16)(x1 * c - x2 * s);
        k[o + 64] = (bf16)(x2 * c + x1 * s);
    }
}

// V transpose: vT (B,G,HS,T) so columns of V are contiguous for B fragments.
__global__ void v_transpose_kernel(const float* __restrict__ qkv,
                                   bf16* __restrict__ vT) {
    int tid = blockIdx.x * blockDim.x + threadIdx.x;
    const int hs = tid & (HSc - 1);
    int rest = tid >> 7;
    const int t = rest % Tc;
    rest /= Tc;
    const int g = rest % Gc;
    const int b = rest / Gc;
    if (b >= Bc) return;
    const float v = qkv[(size_t)(b * Tc + t) * QKVO + (Hc + Gc) * HSc + g * HSc + hs];
    vT[(((size_t)b * Gc + g) * HSc + hs) * Tc + t] = (bf16)v;
}

// ---------------- GEMM: C[M,N] = act(A[M,K](bf16) * W[N,K]^T(bf16) + res) ----------------
// Block: 256 threads = 8 waves as 2(M) x 4(N); block tile 32M x 256N; wave tile 16M x 64N.
__global__ __launch_bounds__(256)
void gemm_bf16_kernel(const bf16* __restrict__ A, const bf16* __restrict__ W,
                      const float* __restrict__ res, float* __restrict__ outF,
                      bf16* __restrict__ outH, int M, int N, int K, int act_gelu) {
    const int lane = threadIdx.x & 31;
    const int wid  = threadIdx.x >> 5;
    const int wm = wid & 1, wn = wid >> 1;
    const int m0 = blockIdx.y * 32 + wm * 16;
    const int n0 = blockIdx.x * 256 + wn * 64;

    const bf16* arow  = A + (size_t)(m0 + (lane & 15)) * K;
    const bf16* wrow0 = W + (size_t)(n0 +  0 + (lane & 15)) * K;
    const bf16* wrow1 = W + (size_t)(n0 + 16 + (lane & 15)) * K;
    const bf16* wrow2 = W + (size_t)(n0 + 32 + (lane & 15)) * K;
    const bf16* wrow3 = W + (size_t)(n0 + 48 + (lane & 15)) * K;

    v8f c0 = {}, c1 = {}, c2 = {}, c3 = {};
    for (int k = 0; k < K; k += 32) {
        __builtin_prefetch(arow + k + 128, 0, 3);
        __builtin_prefetch(wrow0 + k + 128, 0, 3);
        v16bf a  = load_a_frag(arow, lane, k);
        v16bf b0 = load_b_frag(wrow0, lane, k);
        v16bf b1 = load_b_frag(wrow1, lane, k);
        v16bf b2 = load_b_frag(wrow2, lane, k);
        v16bf b3 = load_b_frag(wrow3, lane, k);
        c0 = WMMA_BF16(a, b0, c0);
        c1 = WMMA_BF16(a, b1, c1);
        c2 = WMMA_BF16(a, b2, c2);
        c3 = WMMA_BF16(a, b3, c3);
    }

    // C/D layout: vgpr r, lanes 0-15 -> row m0+r, lanes 16-31 -> row m0+r+8; col = n0+lane%16.
    v8f cc[4] = {c0, c1, c2, c3};
    const int rbase = m0 + ((lane >> 4) & 1) * 8;
    const int cbase = n0 + (lane & 15);
#pragma unroll
    for (int j = 0; j < 4; ++j) {
#pragma unroll
        for (int r = 0; r < 8; ++r) {
            const int row = rbase + r;
            const int col = cbase + j * 16;
            const size_t o = (size_t)row * N + col;
            float v = cc[j][r];
            if (res) v += res[o];
            if (act_gelu) {
                const float t = 0.7978845608028654f * (v + 0.044715f * v * v * v);
                v = 0.5f * v * (1.0f + tanhf(t));
            }
            if (outF) outF[o] = v;
            if (outH) outH[o] = (bf16)v;
        }
    }
}

// ---------------- Flash attention (causal GQA) ----------------
// One wave per 16-row query tile; 8 waves/block. Online softmax with WMMA.
__global__ __launch_bounds__(256)
void attn_kernel(const bf16* __restrict__ q, const bf16* __restrict__ kk,
                 const bf16* __restrict__ vT, bf16* __restrict__ yout) {
    __shared__ bf16 ptile[8][16 * 32];
    const int lane = threadIdx.x & 31;
    const int wid  = threadIdx.x >> 5;
    const int tile = blockIdx.x * 8 + wid;
    const int mt = tile % (Tc / 16);
    const int bh = tile / (Tc / 16);
    const int h = bh % Hc, b = bh / Hc;
    const int g = h / (Hc / Gc);
    const int m0 = mt * 16;
    const float scale = 0.08838834764831845f; // 1/sqrt(128)

    const bf16* qrow = q + (((size_t)b * Hc + h) * Tc + m0 + (lane & 15)) * HSc;
    v16bf qa[4];
#pragma unroll
    for (int kc = 0; kc < 4; ++kc) qa[kc] = load_a_frag(qrow, lane, kc * 32);

    const bf16* kbase = kk + ((size_t)b * Gc + g) * Tc * HSc;
    const bf16* vbase = vT + ((size_t)b * Gc + g) * (size_t)HSc * Tc;
    bf16* pt = &ptile[wid][0];

    v8f zero = {};
    v8f yacc[8];
#pragma unroll
    for (int f = 0; f < 8; ++f) yacc[f] = zero;
    float rowmax[8], rowsum[8];
#pragma unroll
    for (int r = 0; r < 8; ++r) { rowmax[r] = -1e30f; rowsum[r] = 0.f; }

    const int colbase = lane & 15;
    const int rhalf = ((lane >> 4) & 1) * 8;

    for (int s0 = 0; s0 <= m0 + 15; s0 += 32) {
        // scores for two 16-key subtiles, K-dim = HS = 128
        v8f sc[2];
#pragma unroll
        for (int j = 0; j < 2; ++j) {
            const int n0 = s0 + j * 16;
            const bf16* krow = kbase + (size_t)(n0 + (lane & 15)) * HSc;
            v8f s = zero;
#pragma unroll
            for (int kc = 0; kc < 4; ++kc) {
                v16bf bfr = load_b_frag(krow, lane, kc * 32);
                s = WMMA_BF16(qa[kc], bfr, s);
            }
            sc[j] = s;
        }

        float alpha[8];
#pragma unroll
        for (int r = 0; r < 8; ++r) {
            const int row = m0 + rhalf + r;
            float v0 = sc[0][r] * scale;
            float v1 = sc[1][r] * scale;
            if (s0 + colbase > row)      v0 = -1e30f;
            if (s0 + 16 + colbase > row) v1 = -1e30f;
            float mx = fmaxf(v0, v1);
            mx = fmaxf(mx, __shfl_xor(mx, 1, 32));
            mx = fmaxf(mx, __shfl_xor(mx, 2, 32));
            mx = fmaxf(mx, __shfl_xor(mx, 4, 32));
            mx = fmaxf(mx, __shfl_xor(mx, 8, 32));
            const float nm = fmaxf(rowmax[r], mx);
            const float al = __expf(rowmax[r] - nm);
            const float p0 = __expf(v0 - nm);
            const float p1 = __expf(v1 - nm);
            float ps = p0 + p1;
            ps += __shfl_xor(ps, 1, 32);
            ps += __shfl_xor(ps, 2, 32);
            ps += __shfl_xor(ps, 4, 32);
            ps += __shfl_xor(ps, 8, 32);
            rowsum[r] = rowsum[r] * al + ps;
            rowmax[r] = nm;
            alpha[r] = al;
            // stage P tile (row-major 16x32) in LDS for A-layout reload
            pt[(rhalf + r) * 32 + colbase]      = (bf16)p0;
            pt[(rhalf + r) * 32 + 16 + colbase] = (bf16)p1;
        }

        // rescale running output accumulators (row mapping matches C layout)
#pragma unroll
        for (int f = 0; f < 8; ++f)
#pragma unroll
            for (int r = 0; r < 8; ++r) yacc[f][r] *= alpha[r];

        // cross-lane LDS dependency within the wave: drain DS before reload
        asm volatile("s_wait_dscnt 0" ::: "memory");
        v16bf pa = load_a_frag(pt + (size_t)(lane & 15) * 32, lane, 0);

#pragma unroll
        for (int f = 0; f < 8; ++f) {
            const bf16* vrow = vbase + (size_t)(f * 16 + (lane & 15)) * Tc;
            v16bf vb = load_b_frag(vrow, lane, s0);
            yacc[f] = WMMA_BF16(pa, vb, yacc[f]);
        }
    }

    // normalize and store y (B,T,H*HS) as bf16 (A matrix of the proj GEMM)
#pragma unroll
    for (int r = 0; r < 8; ++r) {
        const float rin = 1.0f / rowsum[r];
        const int row = m0 + rhalf + r;
        const size_t obase = ((size_t)b * Tc + row) * (Hc * HSc) + (size_t)h * HSc;
#pragma unroll
        for (int f = 0; f < 8; ++f)
            yout[obase + f * 16 + (lane & 15)] = (bf16)(yacc[f][r] * rin);
    }
}

// ---------------- host orchestration ----------------
extern "C" void kernel_launch(void* const* d_in, const int* in_sizes, int n_in,
                              void* d_out, int out_size, void* d_ws, size_t ws_size,
                              hipStream_t stream) {
    (void)in_sizes; (void)n_in; (void)out_size; (void)ws_size;

    const int*   idx         = (const int*)d_in[0];
    const float* wte         = (const float*)d_in[1];
    const float* qkv_w       = (const float*)d_in[2];
    const float* attn_proj_w = (const float*)d_in[3];
    const float* fc_w        = (const float*)d_in[4];
    const float* mlp_proj_w  = (const float*)d_in[5];
    const float* norm1_w     = (const float*)d_in[6];
    const float* norm2_w     = (const float*)d_in[7];
    const float* lnf_w       = (const float*)d_in[8];
    const float* lm_head_w   = (const float*)d_in[9];

    char* ws = (char*)d_ws;
    size_t off = 0;
    auto alloc = [&](size_t bytes) -> void* {
        void* p = ws + off;
        off += (bytes + 255) & ~(size_t)255;
        return p;
    };

    bf16* wqkv  = (bf16*)alloc((size_t)Lc * QKVO * Ec * 2);
    bf16* wproj = (bf16*)alloc((size_t)Lc * Ec * Hc * HSc * 2);
    bf16* wfc   = (bf16*)alloc((size_t)Lc * FFc * Ec * 2);
    bf16* wmlp  = (bf16*)alloc((size_t)Lc * Ec * FFc * 2);
    bf16* wlm   = (bf16*)alloc((size_t)Vc * Ec * 2);
    float* x    = (float*)alloc((size_t)Mr * Ec * 4);
    bf16* xn    = (bf16*)alloc((size_t)Mr * Ec * 2);
    float* qkv  = (float*)alloc((size_t)Mr * QKVO * 4);
    bf16* qb    = (bf16*)alloc((size_t)Bc * Hc * Tc * HSc * 2);
    bf16* kb    = (bf16*)alloc((size_t)Bc * Gc * Tc * HSc * 2);
    bf16* vtb   = (bf16*)alloc((size_t)Bc * Gc * HSc * Tc * 2);
    bf16* yattn = (bf16*)alloc((size_t)Mr * Hc * HSc * 2);
    bf16* hb    = (bf16*)alloc((size_t)Mr * FFc * 2);

    auto cvt = [&](const float* s, bf16* d, long long n) {
        int blocks = (int)((n + 2047) / 2048);
        f32_to_bf16_kernel<<<blocks, 256, 0, stream>>>(s, d, n);
    };
    cvt(qkv_w,       wqkv,  (long long)Lc * QKVO * Ec);
    cvt(attn_proj_w, wproj, (long long)Lc * Ec * Hc * HSc);
    cvt(fc_w,        wfc,   (long long)Lc * FFc * Ec);
    cvt(mlp_proj_w,  wmlp,  (long long)Lc * Ec * FFc);
    cvt(lm_head_w,   wlm,   (long long)Vc * Ec);

    embed_kernel<<<Mr, 256, 0, stream>>>(idx, wte, x);

    const dim3 blk(256);
    for (int l = 0; l < Lc; ++l) {
        rmsnorm_kernel<<<Mr, blk, 0, stream>>>(x, norm1_w + (size_t)l * Ec, xn);

        // qkv: (2048 x 3072 x 2048)
        gemm_bf16_kernel<<<dim3(QKVO / 256, Mr / 32), blk, 0, stream>>>(
            xn, wqkv + (size_t)l * QKVO * Ec, nullptr, qkv, nullptr, Mr, QKVO, Ec, 0);

        rope_qk_kernel<<<(Bc * Tc * (Hc + Gc) * 64) / 256, blk, 0, stream>>>(qkv, qb, kb);
        v_transpose_kernel<<<(Bc * Gc * Tc * HSc) / 256, blk, 0, stream>>>(qkv, vtb);

        attn_kernel<<<(Bc * Hc * (Tc / 16)) / 8, blk, 0, stream>>>(qb, kb, vtb, yattn);

        // attn proj + residual: (2048 x 2048 x 2048)
        gemm_bf16_kernel<<<dim3(Ec / 256, Mr / 32), blk, 0, stream>>>(
            yattn, wproj + (size_t)l * Ec * Hc * HSc, x, x, nullptr, Mr, Ec, Hc * HSc, 0);

        rmsnorm_kernel<<<Mr, blk, 0, stream>>>(x, norm2_w + (size_t)l * Ec, xn);

        // fc + gelu, bf16 out: (2048 x 8192 x 2048)
        gemm_bf16_kernel<<<dim3(FFc / 256, Mr / 32), blk, 0, stream>>>(
            xn, wfc + (size_t)l * FFc * Ec, nullptr, nullptr, hb, Mr, FFc, Ec, 1);

        // mlp proj + residual: (2048 x 2048 x 8192)
        gemm_bf16_kernel<<<dim3(Ec / 256, Mr / 32), blk, 0, stream>>>(
            hb, wmlp + (size_t)l * Ec * FFc, x, x, nullptr, Mr, Ec, FFc, 0);
    }

    rmsnorm_kernel<<<Mr, blk, 0, stream>>>(x, lnf_w, xn);

    // lm head: (2048 x 32000 x 2048) -> fp32 logits
    gemm_bf16_kernel<<<dim3(Vc / 256, Mr / 32), blk, 0, stream>>>(
        xn, wlm, nullptr, (float*)d_out, nullptr, Mr, Vc, Ec, 0);
}
